// AdaptiveAttentionTransformerBlock_38345468019312
// MI455X (gfx1250) — compile-verified, matrix operand order checked by reading, not run on hardware
//
#include <hip/hip_runtime.h>
#include <hip/hip_bf16.h>
#include <math.h>

#define E        1024
#define Hh       16
#define Dd       64
#define Ff       4608
#define BASEFFN  3072
#define Ll       4096
#define Bb       2
#define ROWS     (Bb * Ll)   // 8192
#define CHUNK    256
#define NCH      (Ll / CHUNK)

typedef __attribute__((ext_vector_type(16))) __bf16 v16bf;
typedef __attribute__((ext_vector_type(8)))  float  v8f;

__device__ __forceinline__ unsigned short f2bf(float f) {
  union { float f; unsigned u; } x; x.f = f;
  unsigned r = x.u + 0x7FFFu + ((x.u >> 16) & 1u);
  return (unsigned short)(r >> 16);
}
__device__ __forceinline__ float bf2f(unsigned short h) {
  union { unsigned u; float f; } x; x.u = ((unsigned)h) << 16;
  return x.f;
}

struct Frag { union { v16bf v; uint4 q[2]; unsigned short u[16]; }; };
__device__ __forceinline__ void ldfrag(Frag& f, const unsigned short* p) {
  f.q[0] = *(const uint4*)p;
  f.q[1] = *(const uint4*)(p + 16);
}
__device__ __forceinline__ v8f vzero() {
  v8f z = {0.f,0.f,0.f,0.f,0.f,0.f,0.f,0.f};
  return z;
}
#define WMMA_BF16(A,B,C) \
  __builtin_amdgcn_wmma_f32_16x16x32_bf16(false,(A),false,(B),(short)0,(C),false,false)

// CDNA5 async memory->LDS copy (16B per lane), tracked by ASYNCcnt.
__device__ __forceinline__ void async_g2l_b128(unsigned lds_byte_addr, const void* gaddr) {
  asm volatile("global_load_async_to_lds_b128 %0, %1, off"
               :: "v"(lds_byte_addr), "v"((unsigned long long)gaddr) : "memory");
}
__device__ __forceinline__ unsigned lds_addr32(const void* p) {
  return (unsigned)(unsigned long long)p;   // LDS aperture: low 32 bits = LDS offset
}

// ---------------------------------------------------------------- fp32 -> bf16
__global__ __launch_bounds__(256)
void cvt_f32_bf16_kernel(const float* __restrict__ in, unsigned short* __restrict__ out, int n) {
  int i = blockIdx.x * 256 + threadIdx.x;
  if (i < n) out[i] = f2bf(in[i]);
}

// ---------------------------------------------------------------- RMS norm row
__global__ __launch_bounds__(256)
void rmsnorm_kernel(const float* __restrict__ x, const float* __restrict__ g,
                    unsigned short* __restrict__ outbf) {
  __shared__ float red[256];
  const int row = blockIdx.x, tid = threadIdx.x;
  const size_t base = (size_t)row * E;
  float rv[4]; float ss = 0.f;
#pragma unroll
  for (int it = 0; it < 4; ++it) {
    int col = tid + it * 256;
    float v = x[base + col];
    if (!(v == v) || fabsf(v) > 3.0e38f) v = 0.f;   // nan/inf scrub
    rv[it] = v; ss += v * v;
  }
  red[tid] = ss; __syncthreads();
  for (int s = 128; s > 0; s >>= 1) { if (tid < s) red[tid] += red[tid + s]; __syncthreads(); }
  float rms = sqrtf(red[0] * (1.f / E) + 1e-6f);
  rms = fminf(fmaxf(rms, 1e-6f), 1e6f);
  float inv = 1.f / rms;
#pragma unroll
  for (int it = 0; it < 4; ++it) {
    int col = tid + it * 256;
    outbf[base + col] = f2bf(rv[it] * inv * g[col]);
  }
}

// -------------------------------------------------- generic bf16 WMMA GEMM
// C(MxN) = A(MxK) @ B(KxN)  [+ R], A/B bf16, C fp32 or bf16.
// Double-buffered LDS. A tile streams via global_load_async_to_lds_b128;
// B tile loads to VGPRs at loop top (overlapping WMMAs) and is stored
// transposed after compute, so all fragment loads are contiguous ds_load_b128.
// One barrier per K-step.
__global__ __launch_bounds__(256)
void gemm_bf16_wmma(const unsigned short* __restrict__ A,
                    const unsigned short* __restrict__ B,
                    const float* __restrict__ R,
                    float* __restrict__ Cf,
                    unsigned short* __restrict__ Cbf,
                    int M, int N, int K) {
  __shared__ unsigned short As[2][128][40];   // [m][k], stride 40 (16B-aligned)
  __shared__ unsigned short Bs[2][128][40];   // transposed [n][k]
  const int tid = threadIdx.x;
  const int lane = tid & 31, w = tid >> 5;
  const int half = lane >> 4, l15 = lane & 15;
  const int wm = (w >> 2) * 64, wn = (w & 3) * 32;
  const int bm = blockIdx.y * 128, bn = blockIdx.x * 128;

  v8f acc[4][2];
  for (int i = 0; i < 4; ++i) for (int j = 0; j < 2; ++j) acc[i][j] = vzero();

  const int ar = tid >> 1, ac = (tid & 1) * 16;   // A: 2 threads/row, 16 el each
  const int br = tid >> 3, bc = (tid & 7) * 16;   // B: 8 threads/row, 16 el each
  const unsigned short* gA0 = A + (size_t)(bm + ar) * K + ac;
  const unsigned short* gB0 = B + (size_t)br * N + bn + bc;

  auto issueA = [&](int k0, int bufi) {           // 2 asyncs per lane
    unsigned la = lds_addr32(&As[bufi][ar][ac]);
    const unsigned short* ga = gA0 + k0;
    async_g2l_b128(la,      ga);
    async_g2l_b128(la + 16, ga + 8);
  };
  auto loadB = [&](int k0, uint4& t0, uint4& t1) {
    const unsigned short* gb = gB0 + (size_t)k0 * N;
    t0 = *(const uint4*)gb;
    t1 = *(const uint4*)(gb + 8);
  };
  auto storeB = [&](int bufi, const uint4& t0, const uint4& t1) {
    unsigned short tmp[16];
    *(uint4*)tmp       = t0;
    *(uint4*)(tmp + 8) = t1;
#pragma unroll
    for (int j = 0; j < 16; ++j) Bs[bufi][bc + j][br] = tmp[j];
  };

  // prologue: tile 0
  issueA(0, 0);
  { uint4 t0, t1; loadB(0, t0, t1); storeB(0, t0, t1); }
  asm volatile("s_wait_asynccnt 0x0" ::: "memory");
  __syncthreads();

  int buf = 0;
  for (int k0 = 0; k0 < K; k0 += 32) {
    const bool has_next = (k0 + 32 < K);
    uint4 t0, t1;
    if (has_next) {
      issueA(k0 + 32, buf ^ 1);                   // async A for next tile
      loadB(k0 + 32, t0, t1);                     // B for next tile -> VGPRs
      if (k0 + 64 < K) {                          // L2 prefetch 2 tiles ahead
        __builtin_prefetch(gA0 + (k0 + 64), 0, 1);
        __builtin_prefetch(gB0 + (size_t)(k0 + 64) * N, 0, 1);
      }
    }

    Frag fa[4], fb[2];
#pragma unroll
    for (int i = 0; i < 4; ++i) ldfrag(fa[i], &As[buf][wm + i*16 + l15][half*8]);
#pragma unroll
    for (int j = 0; j < 2; ++j) ldfrag(fb[j], &Bs[buf][wn + j*16 + l15][half*8]);
#pragma unroll
    for (int i = 0; i < 4; ++i)
#pragma unroll
      for (int j = 0; j < 2; ++j)
        acc[i][j] = WMMA_BF16(fa[i].v, fb[j].v, acc[i][j]);

    if (has_next) {
      storeB(buf ^ 1, t0, t1);                    // transpose-store B(next)
      asm volatile("s_wait_asynccnt 0x0" ::: "memory");  // A(next) landed in LDS
    }
    __syncthreads();                              // publish next tile / retire cur
    buf ^= 1;
  }

#pragma unroll
  for (int i = 0; i < 4; ++i)
#pragma unroll
    for (int j = 0; j < 2; ++j) {
      const int row0 = bm + wm + i*16 + (half ? 8 : 0);
      const int col  = bn + wn + j*16 + l15;
#pragma unroll
      for (int r = 0; r < 8; ++r) {
        size_t idx = (size_t)(row0 + r) * N + col;
        float v = acc[i][j][r];
        if (R) v += R[idx];
        if (Cbf) Cbf[idx] = f2bf(v);
        else     Cf[idx]  = v;
      }
    }
}

// ------------------------------------------ RoPE + (elu+1), split qkv -> BHLD
__global__ __launch_bounds__(256)
void rope_elu_kernel(const float* __restrict__ qkv,
                     float* __restrict__ qo, float* __restrict__ ko,
                     float* __restrict__ vo) {
  const int t   = blockIdx.x * 256 + threadIdx.x;   // over B*L*E
  const int b   = t >> 22;
  const int rem = t & ((1 << 22) - 1);
  const int l   = rem >> 10;
  const int e   = rem & 1023;
  const int h   = e >> 6, d = e & 63;
  const size_t rbase = ((size_t)(b * Ll + l)) * (3 * E);
  float qv = qkv[rbase + e];
  float kv = qkv[rbase + E + e];
  float vv = qkv[rbase + 2 * E + e];
  const int j = d & 31;
  const float inv = expf((float)j * (-9.2103403719761836f / 32.f)); // 10000^{-j/32}
  const float ang = (float)l * inv;
  const float cs = cosf(ang), sn = sinf(ang);
  const float qp = __shfl_xor(qv, 1);
  const float kp = __shfl_xor(kv, 1);
  const float sgn = (d & 1) ? 1.f : -1.f;   // rot[2i]=-x[2i+1], rot[2i+1]=x[2i]
  float q2 = qv * cs + sgn * qp * sn;
  float k2 = kv * cs + sgn * kp * sn;
  q2 = (q2 > 0.f) ? (q2 + 1.f) : expf(q2);  // elu(x)+1
  k2 = (k2 > 0.f) ? (k2 + 1.f) : expf(k2);
  const size_t obase = (((size_t)(b * Hh + h)) * Ll + l) * Dd + d;
  qo[obase] = q2; ko[obase] = k2; vo[obase] = vv;
}

// ------------------------------------------------ chunked linear attention
// One block per (b,h).  num = q@S + mask(q k^T)@v  (WMMA),
// den = q.(Z + cumsum k) via masked-A rowsums, S/Z kept fp32 in LDS.
#define ATTN_LDS_BYTES 188928
__global__ __launch_bounds__(256)
void linattn_kernel(const float* __restrict__ qg, const float* __restrict__ kg,
                    const float* __restrict__ vg, unsigned short* __restrict__ outbf) {
  extern __shared__ char smem[];
  unsigned short (*qb)[72]   = (unsigned short(*)[72])(smem);            // 256x72 bf16
  unsigned short (*kb)[72]   = (unsigned short(*)[72])(smem + 36864);    // 256x72
  unsigned short (*kbT)[264] = (unsigned short(*)[264])(smem + 73728);   // 64x264
  unsigned short (*vbT)[264] = (unsigned short(*)[264])(smem + 107520);  // 64x264
  float (*Sf)[65]            = (float(*)[65])(smem + 141312);            // 64x65 f32
  unsigned short (*SbT)[72]  = (unsigned short(*)[72])(smem + 157952);   // 64x72 bf16 (S^T)
  float* Zf                  = (float*)(smem + 167168);                  // 64
  float* den                 = (float*)(smem + 167424);                  // 256

  const int bh = blockIdx.x;
  const int b = bh >> 4, h = bh & 15;
  const int tid = threadIdx.x;
  const int lane = tid & 31, w = tid >> 5;
  const int half = lane >> 4, l15 = lane & 15;
  unsigned short (*Am)[40]   = (unsigned short(*)[40])(smem + 168448 + w * 2560); // 32x40/wave

  for (int i = tid; i < 64 * 65; i += 256) ((float*)Sf)[i] = 0.f;
  for (int i = tid; i < 64 * 72; i += 256) ((unsigned short*)SbT)[i] = 0;
  if (tid < 64) Zf[tid] = 0.f;
  __syncthreads();

  const int rowbase = w * 32;

  for (int c = 0; c < NCH; ++c) {
    const size_t base = ((size_t)bh * Ll + c * CHUNK) * Dd;
    for (int e = tid; e < CHUNK * Dd; e += 256) {
      const int pos = e >> 6, d = e & 63;
      qb[pos][d] = f2bf(qg[base + e]);
      unsigned short kk = f2bf(kg[base + e]);
      kb[pos][d] = kk;
      kbT[d][pos] = kk;
      vbT[d][pos] = f2bf(vg[base + e]);
    }
    __syncthreads();

    { // den init: q . Z_prev
      float s = 0.f;
      for (int d = 0; d < 64; ++d) s += bf2f(qb[tid][d]) * Zf[d];
      den[tid] = s;
    }
    __syncthreads();

    v8f acc[2][4];
    for (int mt = 0; mt < 2; ++mt) for (int nt = 0; nt < 4; ++nt) acc[mt][nt] = vzero();

    // inter-chunk: q @ S_prev   (B operand from transposed bf16 S)
#pragma unroll
    for (int ks = 0; ks < 2; ++ks) {
      Frag fq[2], fs[4];
#pragma unroll
      for (int mt = 0; mt < 2; ++mt)
        ldfrag(fq[mt], &qb[rowbase + mt*16 + l15][ks*32 + half*8]);
#pragma unroll
      for (int nt = 0; nt < 4; ++nt)
        ldfrag(fs[nt], &SbT[nt*16 + l15][ks*32 + half*8]);
#pragma unroll
      for (int mt = 0; mt < 2; ++mt)
#pragma unroll
        for (int nt = 0; nt < 4; ++nt)
          acc[mt][nt] = WMMA_BF16(fq[mt].v, fs[nt].v, acc[mt][nt]);
    }

    // intra-chunk causal: only column blocks j <= w survive masking
    for (int j = 0; j <= w; ++j) {
      v8f at[2][2];
      for (int mt = 0; mt < 2; ++mt) for (int ct = 0; ct < 2; ++ct) at[mt][ct] = vzero();
#pragma unroll
      for (int ks = 0; ks < 2; ++ks) {
        Frag fq[2], fk[2];
#pragma unroll
        for (int mt = 0; mt < 2; ++mt)
          ldfrag(fq[mt], &qb[rowbase + mt*16 + l15][ks*32 + half*8]);
#pragma unroll
        for (int ct = 0; ct < 2; ++ct)
          ldfrag(fk[ct], &kb[j*32 + ct*16 + l15][ks*32 + half*8]);
#pragma unroll
        for (int mt = 0; mt < 2; ++mt)
#pragma unroll
          for (int ct = 0; ct < 2; ++ct)
            at[mt][ct] = WMMA_BF16(fq[mt].v, fk[ct].v, at[mt][ct]);
      }
      const bool diag = (j == w);
#pragma unroll
      for (int mt = 0; mt < 2; ++mt)
#pragma unroll
        for (int ct = 0; ct < 2; ++ct) {
#pragma unroll
          for (int r = 0; r < 8; ++r) {
            const int rl  = mt*16 + r + (half ? 8 : 0);
            const int row = rowbase + rl;
            const int col = j*32 + ct*16 + l15;
            float v = at[mt][ct][r];
            if (diag && col > row) v = 0.f;       // causal mask (m <= l)
            float s = v;                           // rowsum -> den
            s += __shfl_xor(s, 1);
            s += __shfl_xor(s, 2);
            s += __shfl_xor(s, 4);
            s += __shfl_xor(s, 8);
            if (l15 == 0) den[row] += s;
            Am[rl][ct*16 + l15] = f2bf(v);         // stage masked A as bf16
          }
        }
      Frag fa[2], fv[4];
#pragma unroll
      for (int mt = 0; mt < 2; ++mt) ldfrag(fa[mt], &Am[mt*16 + l15][half*8]);
#pragma unroll
      for (int nt = 0; nt < 4; ++nt) ldfrag(fv[nt], &vbT[nt*16 + l15][j*32 + half*8]);
#pragma unroll
      for (int mt = 0; mt < 2; ++mt)
#pragma unroll
        for (int nt = 0; nt < 4; ++nt)
          acc[mt][nt] = WMMA_BF16(fa[mt].v, fv[nt].v, acc[mt][nt]);
    }

    // normalized output -> (B, L, H*D) bf16
#pragma unroll
    for (int mt = 0; mt < 2; ++mt)
#pragma unroll
      for (int nt = 0; nt < 4; ++nt) {
#pragma unroll
        for (int r = 0; r < 8; ++r) {
          const int row = rowbase + mt*16 + r + (half ? 8 : 0);
          const int d   = nt*16 + l15;
          const float dv = fmaxf(den[row], 1e-6f);
          const float o  = acc[mt][nt][r] / dv;
          const int pos = c * CHUNK + row;
          outbf[((size_t)b * Ll + pos) * E + h * Dd + d] = f2bf(o);
        }
      }
    __syncthreads();

    // state update: S += k^T @ v  (2 tiles per wave), Z += colsum(k)
#pragma unroll
    for (int t2 = 0; t2 < 2; ++t2) {
      const int idx = w * 2 + t2;
      const int mt = idx >> 2, nt = idx & 3;
      v8f sa = vzero();
#pragma unroll
      for (int ks = 0; ks < 8; ++ks) {
        Frag fk, fv;
        ldfrag(fk, &kbT[mt*16 + l15][ks*32 + half*8]);
        ldfrag(fv, &vbT[nt*16 + l15][ks*32 + half*8]);
        sa = WMMA_BF16(fk.v, fv.v, sa);
      }
#pragma unroll
      for (int r = 0; r < 8; ++r)
        Sf[mt*16 + r + (half ? 8 : 0)][nt*16 + l15] += sa[r];
    }
    if (tid < 64) {
      float z = Zf[tid];
      for (int p = 0; p < CHUNK; ++p) z += bf2f(kbT[tid][p]);
      Zf[tid] = z;
    }
    __syncthreads();
    for (int i = tid; i < 64 * 64; i += 256) {   // refresh bf16 S^T
      const int dd = i >> 6, ee = i & 63;
      SbT[ee][dd] = f2bf(Sf[dd][ee]);
    }
    __syncthreads();
  }
}

// ---------------------------------------------------------------- x_mean
__global__ __launch_bounds__(256)
void xmean_kernel(const unsigned short* __restrict__ h2, float* __restrict__ xm) {
  int t = blockIdx.x * 256 + threadIdx.x;   // 0 .. B*E-1
  if (t >= Bb * E) return;
  int b = t >> 10, e = t & 1023;
  float s = 0.f;
  const unsigned short* p = h2 + (size_t)b * Ll * E + e;
  for (int l = 0; l < Ll; ++l) s += bf2f(p[(size_t)l * E]);
  xm[t] = s * (1.f / Ll);
}

// ------------------------------------------------- adaptive size MLP (tiny)
__global__ __launch_bounds__(256)
void dp_size_kernel(const float* __restrict__ xm, const float* __restrict__ wdp1,
                    const float* __restrict__ wdp2, int* __restrict__ sizebuf) {
  __shared__ float t1[2][256];
  __shared__ float ratio[2];
  int c = threadIdx.x;
  for (int b = 0; b < 2; ++b) {
    float s = 0.f;
    for (int e = 0; e < E; ++e) s += xm[b * E + e] * wdp1[(size_t)e * 256 + c];
    t1[b][c] = s / (1.f + expf(-s));   // silu
  }
  __syncthreads();
  if (c < 2) {
    float s = 0.f;
    for (int j = 0; j < 256; ++j) s += t1[c][j] * wdp2[j];
    float sig = 1.f / (1.f + expf(-s));
    float r = 1.f + (sig - 0.5f) * 2.f * 0.5f;
    ratio[c] = fminf(fmaxf(r, 0.5f), 1.5f);
  }
  __syncthreads();
  if (c == 0) {
    float rm = 0.5f * (ratio[0] + ratio[1]);
    int sz = (int)floorf((float)BASEFFN * rm);
    if (sz < 1) sz = 1;
    sizebuf[0] = sz;
  }
}

// --------------------------------------- FFN epilogue: mask + masked RMS
__global__ __launch_bounds__(256)
void ffn_epilogue_kernel(const unsigned short* __restrict__ gate,
                         const unsigned short* __restrict__ up,
                         const float* __restrict__ ghid,
                         const int* __restrict__ sizebuf,
                         unsigned short* __restrict__ hiddenbf) {
  __shared__ float red[256];
  const int row = blockIdx.x, tid = threadIdx.x;
  const int size = sizebuf[0];
  const size_t base = (size_t)row * Ff;
  float hv[18];
  float ss = 0.f;
#pragma unroll
  for (int it = 0; it < 18; ++it) {
    int col = tid + it * 256;
    float g = bf2f(gate[base + col]);
    float u = bf2f(up[base + col]);
    g = g / (1.f + expf(-g));
    float hval = g * u;
    if (col >= size) hval = 0.f;
    if (!(hval == hval) || fabsf(hval) > 3.0e38f) hval = 0.f;
    hv[it] = hval;
    ss += hval * hval;
  }
  red[tid] = ss; __syncthreads();
  for (int s = 128; s > 0; s >>= 1) { if (tid < s) red[tid] += red[tid + s]; __syncthreads(); }
  float rms = sqrtf(red[0] / (float)size + 1e-6f);
  rms = fminf(fmaxf(rms, 1e-6f), 1e6f);
  float inv = 1.f / rms;
#pragma unroll
  for (int it = 0; it < 18; ++it) {
    int col = tid + it * 256;
    float o = hv[it] * inv * ghid[col];
    if (col >= size) o = 0.f;
    hiddenbf[base + col] = f2bf(o);
  }
}

// =========================================================== host pipeline
extern "C" void kernel_launch(void* const* d_in, const int* in_sizes, int n_in,
                              void* d_out, int out_size, void* d_ws, size_t ws_size,
                              hipStream_t stream) {
  (void)in_sizes; (void)n_in; (void)out_size; (void)ws_size;
  const float* x      = (const float*)d_in[0];
  const float* w_qkv  = (const float*)d_in[1];
  const float* w_out  = (const float*)d_in[2];
  const float* g1     = (const float*)d_in[3];
  const float* g2     = (const float*)d_in[4];
  const float* w_dp1  = (const float*)d_in[5];
  const float* w_dp2  = (const float*)d_in[6];
  const float* w_gate = (const float*)d_in[7];
  const float* w_up   = (const float*)d_in[8];
  const float* w_down = (const float*)d_in[9];
  const float* g_hid  = (const float*)d_in[10];
  float* out = (float*)d_out;
  char*  ws  = (char*)d_ws;

  size_t o = 0;
  auto alloc = [&](size_t bytes) { size_t r = o; o += (bytes + 255) & ~(size_t)255; return r; };
  const size_t o_wqkv  = alloc((size_t)E * 3 * E * 2);
  const size_t o_wout  = alloc((size_t)E * E * 2);
  const size_t o_wgate = alloc((size_t)E * Ff * 2);
  const size_t o_wup   = alloc((size_t)E * Ff * 2);
  const size_t o_wdown = alloc((size_t)Ff * E * 2);
  const size_t o_x2    = alloc((size_t)ROWS * E * 4);
  const size_t o_h2bf  = alloc((size_t)ROWS * E * 2);
  const size_t o_xmean = alloc((size_t)Bb * E * 4);
  const size_t o_size  = alloc(256);
  const size_t o_scr   = alloc((size_t)234881024);   // union scratch region

  unsigned short* wqkv_bf  = (unsigned short*)(ws + o_wqkv);
  unsigned short* wout_bf  = (unsigned short*)(ws + o_wout);
  unsigned short* wgate_bf = (unsigned short*)(ws + o_wgate);
  unsigned short* wup_bf   = (unsigned short*)(ws + o_wup);
  unsigned short* wdown_bf = (unsigned short*)(ws + o_wdown);
  float*          x2       = (float*)(ws + o_x2);
  unsigned short* h2bf     = (unsigned short*)(ws + o_h2bf);
  float*          xmeanf   = (float*)(ws + o_xmean);
  int*            sizebuf  = (int*)(ws + o_size);
  // phase-1 layout inside scratch
  unsigned short* h_bf   = (unsigned short*)(ws + o_scr);                        // 16.8 MB
  float*          qkvf   = (float*)(ws + o_scr + 16777216);                      // 100.7 MB
  float*          qf     = (float*)(ws + o_scr + 117440512);                     // 33.6 MB
  float*          kf     = (float*)(ws + o_scr + 150994944);                     // 33.6 MB
  float*          vf     = (float*)(ws + o_scr + 184549376);                     // 33.6 MB
  unsigned short* attnbf = (unsigned short*)(ws + o_scr + 218103808);            // 16.8 MB
  // phase-2 layout (aliases phase-1, lifetimes disjoint)
  unsigned short* gatebf = (unsigned short*)(ws + o_scr);                        // 75.5 MB
  unsigned short* upbf   = (unsigned short*)(ws + o_scr + 75497472);             // 75.5 MB
  unsigned short* hidbf  = (unsigned short*)(ws + o_scr + 150994944);            // 75.5 MB

  auto cvt = [&](const float* src, unsigned short* dst, int n) {
    cvt_f32_bf16_kernel<<<(n + 255) / 256, 256, 0, stream>>>(src, dst, n);
  };
  cvt(w_qkv,  wqkv_bf,  E * 3 * E);
  cvt(w_out,  wout_bf,  E * E);
  cvt(w_gate, wgate_bf, E * Ff);
  cvt(w_up,   wup_bf,   E * Ff);
  cvt(w_down, wdown_bf, Ff * E);

  // --- attention branch ---
  rmsnorm_kernel<<<ROWS, 256, 0, stream>>>(x, g1, h_bf);
  gemm_bf16_wmma<<<dim3(3 * E / 128, ROWS / 128), 256, 0, stream>>>(
      h_bf, wqkv_bf, nullptr, qkvf, nullptr, ROWS, 3 * E, E);
  rope_elu_kernel<<<(Bb * Ll * E) / 256, 256, 0, stream>>>(qkvf, qf, kf, vf);
  linattn_kernel<<<Bb * Hh, 256, ATTN_LDS_BYTES, stream>>>(qf, kf, vf, attnbf);
  gemm_bf16_wmma<<<dim3(E / 128, ROWS / 128), 256, 0, stream>>>(
      attnbf, wout_bf, x, x2, nullptr, ROWS, E, E);          // x2 = x + attn@Wout

  // --- adaptive FFN branch ---
  rmsnorm_kernel<<<ROWS, 256, 0, stream>>>(x2, g2, h2bf);
  xmean_kernel<<<(Bb * E + 255) / 256, 256, 0, stream>>>(h2bf, xmeanf);
  dp_size_kernel<<<1, 256, 0, stream>>>(xmeanf, w_dp1, w_dp2, sizebuf);
  gemm_bf16_wmma<<<dim3(Ff / 128, ROWS / 128), 256, 0, stream>>>(
      h2bf, wgate_bf, nullptr, nullptr, gatebf, ROWS, Ff, E);
  gemm_bf16_wmma<<<dim3(Ff / 128, ROWS / 128), 256, 0, stream>>>(
      h2bf, wup_bf, nullptr, nullptr, upbf, ROWS, Ff, E);
  ffn_epilogue_kernel<<<ROWS, 256, 0, stream>>>(gatebf, upbf, g_hid, sizebuf, hidbf);
  gemm_bf16_wmma<<<dim3(E / 128, ROWS / 128), 256, 0, stream>>>(
      hidbf, wdown_bf, x2, out, nullptr, ROWS, E, Ff);       // out = x2 + hidden@Wdown
}
